// complexEmotionPrediction_48799418417237
// MI455X (gfx1250) — matrix operand order repaired and checked by reading, not compiled, not run on hardware
//
#include <hip/hip_runtime.h>

#define E_  30
#define NB_ 12
#define F_  1024
#define H_  32
#define L_  128

typedef __attribute__((ext_vector_type(16))) _Float16 v16h;
typedef __attribute__((ext_vector_type(8)))  float    v8f;
typedef __attribute__((ext_vector_type(4)))  int      v4i;

union Frag16 {
    v16h v;
    unsigned u[8];
    _Float16 h[16];
};

union H2 {
    _Float16 h[2];
    unsigned u;
};

#if __has_builtin(__builtin_amdgcn_global_load_async_to_lds_b128) && \
    __has_builtin(__builtin_amdgcn_s_wait_asynccnt)
#define HAVE_ASYNC_LDS 1
#else
#define HAVE_ASYNC_LDS 0
#endif

#if HAVE_ASYNC_LDS
typedef __attribute__((address_space(1))) v4i GV4;
typedef __attribute__((address_space(3))) v4i LV4;
__device__ __forceinline__ void async_g2l_b128(const void* g, void* l) {
    __builtin_amdgcn_global_load_async_to_lds_b128(
        (GV4*)(unsigned long long)g,
        (LV4*)(unsigned)(unsigned long long)l,
        0, 0);
}
#define ASYNC_WAIT(n) __builtin_amdgcn_s_wait_asynccnt(n)
#else
#define ASYNC_WAIT(n)
#endif

// Branchless SELU: exp(min(x,0)) - 1 == 0 for x >= 0, so no exec-mask branches.
__device__ __forceinline__ float selu_f(float x) {
    const float scale = 1.0507009873554805f;
    const float sa    = 1.0507009873554805f * 1.6732632423543772f;
    return scale * fmaxf(x, 0.f) + sa * (__expf(fminf(x, 0.f)) - 1.f);
}

// Issue one 128x32 fp32 A-tile into LDS (4 x b128 async per thread -> 4 per wave).
__device__ __forceinline__ void issueA(const float* __restrict__ X, long rowBase,
                                       int k0, float* buf, int tid)
{
    #pragma unroll
    for (int i = 0; i < 4; ++i) {
        const int slot = tid + i * 256;          // 1024 float4 slots
        const int row  = slot >> 3;
        const int kq   = (slot & 7) * 4;
        const float* gsrc = X + (rowBase + row) * F_ + k0 + kq;
        float* ldst = buf + row * 32 + kq;
#if HAVE_ASYNC_LDS
        async_g2l_b128(gsrc, ldst);
#else
        *(float4*)ldst = *(const float4*)gsrc;
#endif
    }
}

// One K-step: build fragments (ISA 16-bit A 16x32 / B 32x16 layouts), 2 WMMAs.
__device__ __forceinline__ void mm_tile(const float* __restrict__ Abuf,
                                        const _Float16* __restrict__ lBall,
                                        int k0, int Mbase, int ln, int hl,
                                        v8f& acc0, v8f& acc1)
{
    const unsigned* lB32 = (const unsigned*)lBall;
    Frag16 a, bb0, bb1;
    #pragma unroll
    for (int v = 0; v < 8; ++v) {
        const int ka = ((v < 4) ? 2 * v : 16 + 2 * (v - 4)) + hl * 8;
        const float2 two = *(const float2*)&Abuf[(Mbase + ln) * 32 + ka];
        H2 p;
        p.h[0] = (_Float16)two.x;
        p.h[1] = (_Float16)two.y;
        a.u[v] = p.u;
        const int kb = ((v < 4) ? 2 * v : 16 + 2 * (v - 4)) + hl * 8;
        bb0.u[v] = lB32[(ln * F_ + k0 + kb) >> 1];
        bb1.u[v] = lB32[((16 + ln) * F_ + k0 + kb) >> 1];
    }
    acc0 = __builtin_amdgcn_wmma_f32_16x16x32_f16(false, a.v, false, bb0.v,
                                                  (short)0, acc0, false, false);
    acc1 = __builtin_amdgcn_wmma_f32_16x16x32_f16(false, a.v, false, bb1.v,
                                                  (short)0, acc1, false, false);
}

// Kernel 1: per (m-tile, emotion) block.
//   acc = X(128x1024) * W1[e](1024x32)  via v_wmma_f32_16x16x32_f16,
//         A tile double-buffered with GLOBAL_LOAD_ASYNC_TO_LDS_B128 (ASYNCcnt pipeline)
//   h   = selu(acc + b1[e]) -> f16 LDS
//   mlp = h * W2[e] (32x12 padded to 16) via one more WMMA
//   upd = allW[e] + mlp + b2[e];  w = upd / max(sum_n |upd|, eps) -> ws (B,E,16)
__global__ __launch_bounds__(256)
void k1_mlp_weights(const float* __restrict__ X,
                    const float* __restrict__ W1,
                    const float* __restrict__ b1,
                    const float* __restrict__ W2,
                    const float* __restrict__ b2,
                    const float* __restrict__ bew,
                    float* __restrict__ wout)
{
    __shared__ float    lAf[2][128 * 32];   // A tile ping-pong, fp32, 2 x 16KB
    __shared__ _Float16 lBall[H_ * F_];     // full W1[e] transposed [h][f] f16, 64KB

    const int tid   = threadIdx.x;
    const int lane  = tid & 31;
    const int wid   = tid >> 5;
    const int hl    = lane >> 4;   // lane half
    const int ln    = lane & 15;
    const int Mbase = wid * 16;
    const int e     = blockIdx.y;
    const long rowBase = (long)blockIdx.x * 128;

    v8f acc0 = {};
    v8f acc1 = {};

    // --- prologue: stage ALL of W1[e] (1024x32) as f16, transposed [h][f] ---
    for (int i = tid; i < (F_ * H_) / 4; i += 256) {   // 8192 float4 slots
        const int f  = i >> 3;
        const int hq = (i & 7) * 4;
        const float4 wv = *(const float4*)(W1 + (size_t)e * F_ * H_ + (size_t)f * H_ + hq);
        lBall[(hq + 0) * F_ + f] = (_Float16)wv.x;
        lBall[(hq + 1) * F_ + f] = (_Float16)wv.y;
        lBall[(hq + 2) * F_ + f] = (_Float16)wv.z;
        lBall[(hq + 3) * F_ + f] = (_Float16)wv.w;
    }
    // --- prologue: issue async loads for A tile 0 ---
    issueA(X, rowBase, 0, lAf[0], tid);

    // --- pipelined K loop: issue t+1, retire t (asynccnt<=4), compute t ---
    for (int kt = 0; kt < 31; ++kt) {
        issueA(X, rowBase, (kt + 1) * 32, lAf[(kt + 1) & 1], tid);
        ASYNC_WAIT(4);          // tile kt's 4 loads done; kt+1's still in flight
        __syncthreads();
        mm_tile(lAf[kt & 1], lBall, kt * 32, Mbase, ln, hl, acc0, acc1);
        __syncthreads();        // readers done before buffer is re-issued next iter
    }
    ASYNC_WAIT(0);              // last tile
    __syncthreads();
    mm_tile(lAf[1], lBall, 31 * 32, Mbase, ln, hl, acc0, acc1);
    __syncthreads();            // protect LDS reuse below

    // --- epilogue part 1: selu(acc + b1) -> f16 h-act in lAf[0], [row][h] ---
    _Float16* lAh = (_Float16*)lAf[0];
    _Float16* lW2 = (_Float16*)lAf[1];      // W2^T padded [n][k], n in [0,16)
    const float b1v0 = b1[e * H_ + ln];
    const float b1v1 = b1[e * H_ + 16 + ln];
    #pragma unroll
    for (int r = 0; r < 8; ++r) {
        const int row = Mbase + r + 8 * hl;  // C/D layout: M = r + 8*(lane/16)
        lAh[row * 32 + ln]      = (_Float16)selu_f(acc0[r] + b1v0);
        lAh[row * 32 + 16 + ln] = (_Float16)selu_f(acc1[r] + b1v1);
    }
    for (int i = tid; i < 512; i += 256) {
        const int n  = i >> 5;
        const int hk = i & 31;
        const float wv = (n < NB_) ? W2[((size_t)e * H_ + hk) * NB_ + n] : 0.f;
        lW2[n * 32 + hk] = (_Float16)wv;
    }
    __syncthreads();

    // --- second WMMA: (16x32 h-act) * (32x16 W2pad) ---
    const unsigned* lA32u = (const unsigned*)lAh;
    const unsigned* lW32u = (const unsigned*)lW2;
    Frag16 a2, bw;
    #pragma unroll
    for (int v = 0; v < 8; ++v) {
        const int ka = ((v < 4) ? 2 * v : 16 + 2 * (v - 4)) + hl * 8;
        a2.u[v] = lA32u[((Mbase + ln) * 32 + ka) >> 1];
        const int kb = ((v < 4) ? 2 * v : 16 + 2 * (v - 4)) + hl * 8;
        bw.u[v] = lW32u[(ln * 32 + kb) >> 1];
    }
    v8f acc2 = {};
    acc2 = __builtin_amdgcn_wmma_f32_16x16x32_f16(false, a2.v, false, bw.v,
                                                  (short)0, acc2, false, false);

    // --- allW[e][n] = exp(bew)/sum(exp), b2 ---
    float ew_n = 0.f, ew_sum = 0.f;
    #pragma unroll
    for (int j = 0; j < NB_; ++j) {
        const float ev = __expf(bew[e * NB_ + j]);
        ew_sum += ev;
        if (j == ln) ew_n = ev;
    }
    const float allw = (ln < NB_) ? ew_n / fmaxf(ew_sum, 1e-12f) : 0.f;
    const float b2v  = (ln < NB_) ? b2[e * NB_ + ln] : 0.f;

    // --- signed L1 normalization over NB (16-lane shuffle reduce), store w ---
    #pragma unroll
    for (int r = 0; r < 8; ++r) {
        const float upd = acc2[r] + b2v + allw;  // == 0 for padded n >= 12
        float s = fabsf(upd);
        s += __shfl_xor(s, 1);
        s += __shfl_xor(s, 2);
        s += __shfl_xor(s, 4);
        s += __shfl_xor(s, 8);
        const float w = upd / fmaxf(s, 1e-12f);
        const int row = Mbase + r + 8 * hl;
        wout[((rowBase + row) * (long)E_ + e) * 16 + ln] = w;
    }
}

// Kernel 2: per batch row b — read D[b] once (async to LDS), mix for all 30
// emotions, fully in-wave softmax over L=128, stream the 126MB output.
__global__ __launch_bounds__(256)
void k2_mix_softmax(const float* __restrict__ D,
                    const float* __restrict__ wk,
                    float* __restrict__ out, int B)
{
    __shared__ float sD[NB_ * L_];  // 6KB
    __shared__ float sW[E_ * 16];   // 1.875KB
    const int tid = threadIdx.x;
    const long b  = blockIdx.x;

#if HAVE_ASYNC_LDS
    // 1536 floats = 384 b128 chunks
    for (int i = tid; i < 384; i += 256)
        async_g2l_b128(D + b * (NB_ * L_) + i * 4, &sD[i * 4]);
#else
    for (int i = tid; i < NB_ * L_; i += 256) sD[i] = D[b * (NB_ * L_) + i];
#endif
    for (int i = tid; i < E_ * 16; i += 256) sW[i] = wk[b * (E_ * 16) + i];
    ASYNC_WAIT(0);
    __syncthreads();

    const int lane = tid & 31;
    const int wid  = tid >> 5;

    for (int e = wid; e < E_; e += 8) {
        float m0 = 0.f, m1 = 0.f, m2 = 0.f, m3 = 0.f;
        #pragma unroll
        for (int n = 0; n < NB_; ++n) {
            const float wv = sW[e * 16 + n];
            m0 = fmaf(wv, sD[n * L_ + lane],      m0);
            m1 = fmaf(wv, sD[n * L_ + lane + 32], m1);
            m2 = fmaf(wv, sD[n * L_ + lane + 64], m2);
            m3 = fmaf(wv, sD[n * L_ + lane + 96], m3);
        }
        float mx = fmaxf(fmaxf(m0, m1), fmaxf(m2, m3));
        #pragma unroll
        for (int msk = 16; msk >= 1; msk >>= 1) mx = fmaxf(mx, __shfl_xor(mx, msk));
        const float p0 = __expf(m0 - mx), p1 = __expf(m1 - mx);
        const float p2 = __expf(m2 - mx), p3 = __expf(m3 - mx);
        float se = p0 + p1 + p2 + p3;
        #pragma unroll
        for (int msk = 16; msk >= 1; msk >>= 1) se += __shfl_xor(se, msk);
        const float inv = 1.f / se;
        float* o = out + ((long)e * B + b) * L_;
        o[lane]      = p0 * inv;
        o[lane + 32] = p1 * inv;
        o[lane + 64] = p2 * inv;
        o[lane + 96] = p3 * inv;
    }
}

extern "C" void kernel_launch(void* const* d_in, const int* in_sizes, int n_in,
                              void* d_out, int out_size, void* d_ws, size_t ws_size,
                              hipStream_t stream)
{
    const float* D   = (const float*)d_in[0];  // (B, NB, L)
    const float* X   = (const float*)d_in[1];  // (B, F)
    const float* bew = (const float*)d_in[2];  // (E, 1, NB, 1)
    const float* W1  = (const float*)d_in[3];  // (E, F, H)
    const float* b1  = (const float*)d_in[4];  // (E, H)
    const float* W2  = (const float*)d_in[5];  // (E, H, NB)
    const float* b2  = (const float*)d_in[6];  // (E, NB)
    float* out = (float*)d_out;                // (E, B, L)

    const int B = in_sizes[0] / (NB_ * L_);    // 8192
    float* wk = (float*)d_ws;                  // (B, E, 16) f32 scratch

    dim3 g1(B / 128, E_);
    k1_mlp_weights<<<g1, 256, 0, stream>>>(X, W1, b1, W2, b2, bew, wk);
    k2_mix_softmax<<<B, 256, 0, stream>>>(D, wk, out, B);
}
